// AreaAttention_12395275616864
// MI455X (gfx1250) — matrix operand order repaired and checked by reading, not compiled, not run on hardware
//
#include <hip/hip_runtime.h>
#include <hip/hip_bf16.h>
#include <math.h>

// ---------------------------------------------------------------------------
// Area attention for MI455X (gfx1250, wave32, WMMA).
//   Phase P: tiny prep (q->bf16, W1/W2 transpose->bf16, 9 type-bias vectors)
//   Phase A: area features (mean/std of k-areas -> feat bf16; v area sums f32)
//   Phase G: 3 bf16 WMMA GEMMs: MLP layer1 (relu), MLP layer2, q @ k_area^T
//            (APAD = 8704 = multiple of 64 -> no N-tail, branch-free K loop)
//   Phase S: per-query wave: radix-select top-64 logits, sparse weighted sum
// ---------------------------------------------------------------------------

#define BATCH   8
#define NQ      1024
#define DMODEL  256
#define NAREA   8649
#define APAD    8704            // NAREA padded to multiple of 64 (tail-free GEMM)
#define TOPK    64
#define EPSV    1e-6f

typedef __attribute__((ext_vector_type(16))) __bf16 v16bf;
typedef __attribute__((ext_vector_type(8)))  __bf16 v8bf;
typedef __attribute__((ext_vector_type(8)))  float  v8f;

__constant__ int AREA_START[10] = {0,1024,2016,2976,3968,4929,5859,6819,7749,8649};

__device__ __forceinline__ int area_type(int a) {
  int t = 0;
#pragma unroll
  for (int i = 1; i < 9; i++) t += (a >= AREA_START[i]);
  return t;                    // t = ah*3 + aw
}

// ---------------------------------------------------------------------------
// Prep kernels (tiny)
// ---------------------------------------------------------------------------
__global__ void cvt_q_kernel(const float* __restrict__ q, __bf16* __restrict__ qb, long n) {
  long i = (long)blockIdx.x * blockDim.x + threadIdx.x;
  if (i < n) qb[i] = (__bf16)q[i];
}

__global__ void w1t_kernel(const float* __restrict__ W1, __bf16* __restrict__ W1t) {
  int i = blockIdx.x * blockDim.x + threadIdx.x;      // 256 x 512, N-major
  if (i >= 256 * 512) return;
  int n = i >> 9, kk = i & 511;
  W1t[i] = (__bf16)W1[kk * 256 + n];
}

__global__ void w2t_kernel(const float* __restrict__ W2, __bf16* __restrict__ W2t) {
  int i = blockIdx.x * blockDim.x + threadIdx.x;      // 256 x 256, N-major
  if (i >= 256 * 256) return;
  int n = i >> 8, kk = i & 255;
  W2t[i] = (__bf16)W2[kk * 256 + n];
}

// bias2[t][n] = b1[n] + concat(emb_h[ah], emb_w[aw]) . W1[512:768, n]
__global__ void typebias_kernel(const float* __restrict__ emb_h, const float* __restrict__ emb_w,
                                const float* __restrict__ W1, const float* __restrict__ b1,
                                float* __restrict__ tb) {
  int i = blockIdx.x * blockDim.x + threadIdx.x;      // 9 * 256
  if (i >= 9 * 256) return;
  int t = i >> 8, n = i & 255;
  int ah = t / 3, aw = t % 3;
  float acc = b1[n];
  for (int j = 0; j < 128; j++) acc += emb_h[ah * 128 + j] * W1[(512 + j) * 256 + n];
  for (int j = 0; j < 128; j++) acc += emb_w[aw * 128 + j] * W1[(640 + j) * 256 + n];
  tb[i] = acc;
}

// ---------------------------------------------------------------------------
// Area features: direct <=3x3 cell summation (k, v L2-resident).
// feat[(b*APAD+a)*512 + f]      = mean_k   (bf16)
// feat[(b*APAD+a)*512 + 256+f]  = std_k    (bf16)
// v_area[(b*APAD+a)*256 + f]    = sum_v    (f32)
// ---------------------------------------------------------------------------
__global__ void area_feat_kernel(const float* __restrict__ k, const float* __restrict__ v,
                                 __bf16* __restrict__ feat, float* __restrict__ v_area) {
  long idx = (long)blockIdx.x * blockDim.x + threadIdx.x;
  const long total = (long)BATCH * APAD * (DMODEL / 4);
  if (idx >= total) return;
  int f4 = (int)(idx & 63);
  int a  = (int)((idx >> 6) % APAD);
  int b  = (int)(idx / (64L * APAD));
  __bf16* fr = feat   + ((long)b * APAD + a) * (2 * DMODEL) + f4 * 4;
  float*  vr = v_area + ((long)b * APAD + a) * DMODEL       + f4 * 4;
  if (a >= NAREA) {                       // zero pad rows
#pragma unroll
    for (int c = 0; c < 4; c++) { fr[c] = (__bf16)0.f; fr[DMODEL + c] = (__bf16)0.f; vr[c] = 0.f; }
    return;
  }
  int t = area_type(a);
  int ah = t / 3, aw = t % 3;
  int rel = a - AREA_START[t];
  int wdt = 32 - aw;
  int y = rel / wdt, x = rel % wdt;
  float sk[4] = {0,0,0,0}, sk2[4] = {0,0,0,0}, sv[4] = {0,0,0,0};
  for (int hh = 0; hh <= ah; hh++)
    for (int ww = 0; ww <= aw; ww++) {
      int cell = (y + hh) * 32 + (x + ww);
      float4 kv = *(const float4*)(k + ((long)b * NQ + cell) * DMODEL + f4 * 4);
      float4 vv = *(const float4*)(v + ((long)b * NQ + cell) * DMODEL + f4 * 4);
      float kk4[4] = {kv.x, kv.y, kv.z, kv.w};
      float vv4[4] = {vv.x, vv.y, vv.z, vv.w};
#pragma unroll
      for (int c = 0; c < 4; c++) { sk[c] += kk4[c]; sk2[c] += kk4[c] * kk4[c]; sv[c] += vv4[c]; }
    }
  float inv = 1.f / (float)((ah + 1) * (aw + 1));
#pragma unroll
  for (int c = 0; c < 4; c++) {
    float mean = sk[c] * inv;
    float var  = sk2[c] * inv - mean * mean;
    fr[c]          = (__bf16)mean;
    fr[DMODEL + c] = (__bf16)sqrtf(fabsf(var) + EPSV);
    vr[c] = sv[c];
  }
}

// ---------------------------------------------------------------------------
// WMMA fragment loaders (row-major bf16 sources).
// A (16x32, MxK): lane<16 -> M=lane, K in {0..7, 16..23}; lane>=16 -> K {8..15, 24..31}
// B (32x16, KxN): lane<16 -> N=lane, K {0..7, 16..23}; lane>=16 -> N=lane-16, K {8..15, 24..31}
// ---------------------------------------------------------------------------
__device__ __forceinline__ v16bf load_afrag(const __bf16* __restrict__ A, int lda, int k0) {
  int lane = threadIdx.x & 31;
  int r  = lane & 15;
  int kb = k0 + ((lane >> 4) << 3);
  const __bf16* p = A + (size_t)r * lda + kb;
  v8bf lo = *(const v8bf*)p;
  v8bf hi = *(const v8bf*)(p + 16);
  v16bf o;
#pragma unroll
  for (int i = 0; i < 8; i++) { o[i] = lo[i]; o[i + 8] = hi[i]; }
  return o;
}

__device__ __forceinline__ v16bf load_bfrag(const __bf16* __restrict__ Bm, int ldb, int nbase, int k0) {
  int lane = threadIdx.x & 31;
  int n  = lane & 15;
  int kb = k0 + ((lane >> 4) << 3);
  const __bf16* p = Bm + (size_t)(nbase + n) * ldb + kb;
  v8bf lo = *(const v8bf*)p;
  v8bf hi = *(const v8bf*)(p + 16);
  v16bf o;
#pragma unroll
  for (int i = 0; i < 8; i++) { o[i] = lo[i]; o[i + 8] = hi[i]; }
  return o;
}

// ---------------------------------------------------------------------------
// Branch-free bf16 WMMA GEMM: C(M x N) = A(MxK) * B^T (B stored N-major NxK).
// N is a multiple of 64; each wave owns a 16x64 tile (4 accumulators, A-frag
// reused 4x). K and MODE are compile-time -> fully unrolled, zero inner-loop
// control flow, accumulators pinned (wmma dst == src2).
// MODE 0: += typebias[type(row % APAD)][n], relu   (MLP layer 1)
// MODE 1: += bias[n]                                (MLP layer 2)
// MODE 2: no bias                                   (logits, batched blockIdx.y)
// ---------------------------------------------------------------------------
template<int KDIM, int MODE>
__global__ void __launch_bounds__(256)
gemm_wmma_kernel(const __bf16* __restrict__ A, int lda, long sA,
                 const __bf16* __restrict__ Bm, int ldb, long sB,
                 __bf16* __restrict__ C, int ldc, long sC,
                 int mtiles, int ngroups, const float* __restrict__ bias) {
  const int lane = threadIdx.x & 31;
  const int wave = threadIdx.x >> 5;
  long gw = (long)blockIdx.x * (blockDim.x >> 5) + wave;
  if (gw >= (long)mtiles * ngroups) return;        // wave-uniform
  const int mt = (int)(gw / ngroups);
  const int ng = (int)(gw % ngroups);
  A  += (long)blockIdx.y * sA;
  Bm += (long)blockIdx.y * sB;
  C  += (long)blockIdx.y * sC;
  const __bf16* Arow = A + (size_t)mt * 16 * lda;

  v8f acc0, acc1, acc2, acc3;
#pragma unroll
  for (int i = 0; i < 8; i++) { acc0[i] = 0.f; acc1[i] = 0.f; acc2[i] = 0.f; acc3[i] = 0.f; }

#pragma unroll
  for (int k0 = 0; k0 < KDIM; k0 += 32) {
    v16bf af  = load_afrag(Arow, lda, k0);
    v16bf b0  = load_bfrag(Bm, ldb, ng * 64 +  0, k0);
    v16bf b1  = load_bfrag(Bm, ldb, ng * 64 + 16, k0);
    v16bf b2  = load_bfrag(Bm, ldb, ng * 64 + 32, k0);
    v16bf b3  = load_bfrag(Bm, ldb, ng * 64 + 48, k0);
    acc0 = __builtin_amdgcn_wmma_f32_16x16x32_bf16(false, af, false, b0, (short)0, acc0, false, false);
    acc1 = __builtin_amdgcn_wmma_f32_16x16x32_bf16(false, af, false, b1, (short)0, acc1, false, false);
    acc2 = __builtin_amdgcn_wmma_f32_16x16x32_bf16(false, af, false, b2, (short)0, acc2, false, false);
    acc3 = __builtin_amdgcn_wmma_f32_16x16x32_bf16(false, af, false, b3, (short)0, acc3, false, false);
  }

  const int rowoff = (lane >> 4) << 3;
  const int ncol   = lane & 15;
  int types[8];
  if (MODE == 0) {
#pragma unroll
    for (int r = 0; r < 8; r++) types[r] = area_type((mt * 16 + r + rowoff) % APAD);
  }
  v8f* accs[4] = { &acc0, &acc1, &acc2, &acc3 };
#pragma unroll
  for (int j = 0; j < 4; j++) {
    int n = ng * 64 + j * 16 + ncol;
#pragma unroll
    for (int r = 0; r < 8; r++) {
      int m = mt * 16 + r + rowoff;
      float vv = (*accs[j])[r];
      if (MODE == 0)      { vv += bias[types[r] * 256 + n]; vv = fmaxf(vv, 0.f); }
      else if (MODE == 1) { vv += bias[n]; }
      C[(size_t)m * ldc + n] = (__bf16)vv;
    }
  }
}

// ---------------------------------------------------------------------------
// Selection + output. One wave32 per query row.
// Softmax denominator cancels after top-k renorm: out = sum_top exp(l-m)*v / sum_top exp(l-m).
// 2-level 256-bin radix histogram on order-preserving bf16 bit key -> threshold.
// Third pass: ballot + broadcast survivors; 32 lanes x 8 dims accumulate v_area.
// ---------------------------------------------------------------------------
__global__ void __launch_bounds__(128)
select_out_kernel(const __bf16* __restrict__ logits, const float* __restrict__ v_area,
                  float* __restrict__ out) {
  __shared__ unsigned int hist[4][256];
  const int lane = threadIdx.x & 31;
  const int wave = threadIdx.x >> 5;
  const int qi = blockIdx.x * 4 + wave;
  const int b = qi >> 10;
  const int q = qi & 1023;
  const __bf16* logrow = logits + ((long)b * NQ + q) * APAD;

  for (int i = lane; i < 256; i += 32) hist[wave][i] = 0u;
  __syncthreads();

  // pass 1: max + high-byte histogram
  float fmaxv = -1e30f;
  for (int a0 = 0; a0 < NAREA; a0 += 32) {
    int a = a0 + lane;
    if (a < NAREA) {
      __bf16 hv = logrow[a];
      unsigned short bits = __builtin_bit_cast(unsigned short, hv);
      unsigned key = bits ^ ((bits & 0x8000u) ? 0xFFFFu : 0x8000u);
      atomicAdd(&hist[wave][key >> 8], 1u);
      fmaxv = fmaxf(fmaxv, (float)hv);
    }
  }
#pragma unroll
  for (int off = 16; off >= 1; off >>= 1) fmaxv = fmaxf(fmaxv, __shfl_xor(fmaxv, off, 32));
  __syncthreads();

  int hb = 0, need = TOPK;
  if (lane == 0) {
    int cum = 0;
    for (int bin = 255; bin >= 0; bin--) {
      int c = (int)hist[wave][bin];
      if (cum + c >= TOPK) { hb = bin; need = TOPK - cum; break; }
      cum += c;
    }
  }
  hb = __shfl(hb, 0); need = __shfl(need, 0);
  __syncthreads();

  // pass 2: low-byte histogram within boundary bin
  for (int i = lane; i < 256; i += 32) hist[wave][i] = 0u;
  __syncthreads();
  for (int a0 = 0; a0 < NAREA; a0 += 32) {
    int a = a0 + lane;
    if (a < NAREA) {
      __bf16 hv = logrow[a];
      unsigned short bits = __builtin_bit_cast(unsigned short, hv);
      unsigned key = bits ^ ((bits & 0x8000u) ? 0xFFFFu : 0x8000u);
      if ((int)(key >> 8) == hb) atomicAdd(&hist[wave][key & 255u], 1u);
    }
  }
  __syncthreads();
  int lb = 0;
  if (lane == 0) {
    int cum = 0;
    for (int bin = 255; bin >= 0; bin--) {
      cum += (int)hist[wave][bin];
      if (cum >= need) { lb = bin; break; }
    }
  }
  lb = __shfl(lb, 0);
  const unsigned thr = ((unsigned)hb << 8) | (unsigned)lb;

  // pass 3: select, broadcast, sparse-accumulate
  float accv[8] = {0,0,0,0,0,0,0,0};
  float den = 0.f;
  for (int a0 = 0; a0 < NAREA; a0 += 32) {
    int a = a0 + lane;
    bool sel = false; float w = 0.f;
    if (a < NAREA) {
      __bf16 hv = logrow[a];
      unsigned short bits = __builtin_bit_cast(unsigned short, hv);
      unsigned key = bits ^ ((bits & 0x8000u) ? 0xFFFFu : 0x8000u);
      if (key >= thr) { sel = true; w = __expf((float)hv - fmaxv); }
    }
    unsigned long long mask = __ballot(sel);
    while (mask) {
      int src   = (int)__ffsll(mask) - 1;
      int aidx  = __shfl(a, src);
      float lw  = __shfl(w, src);
      const float* vr = v_area + ((long)b * APAD + aidx) * DMODEL + lane * 8;
      float4 p0 = *(const float4*)vr;
      float4 p1 = *(const float4*)(vr + 4);
      accv[0] += lw * p0.x; accv[1] += lw * p0.y; accv[2] += lw * p0.z; accv[3] += lw * p0.w;
      accv[4] += lw * p1.x; accv[5] += lw * p1.y; accv[6] += lw * p1.z; accv[7] += lw * p1.w;
      den += lw;
      mask &= (mask - 1);
    }
  }
  float invd = 1.f / den;
  float* orow = out + ((long)b * NQ + q) * DMODEL + lane * 8;
  *(float4*)orow       = make_float4(accv[0]*invd, accv[1]*invd, accv[2]*invd, accv[3]*invd);
  *(float4*)(orow + 4) = make_float4(accv[4]*invd, accv[5]*invd, accv[6]*invd, accv[7]*invd);
}

// ---------------------------------------------------------------------------
extern "C" void kernel_launch(void* const* d_in, const int* in_sizes, int n_in,
                              void* d_out, int out_size, void* d_ws, size_t ws_size,
                              hipStream_t stream) {
  const float* q    = (const float*)d_in[0];
  const float* k    = (const float*)d_in[1];
  const float* v    = (const float*)d_in[2];
  const float* embh = (const float*)d_in[3];
  const float* embw = (const float*)d_in[4];
  const float* W1   = (const float*)d_in[5];
  const float* b1   = (const float*)d_in[6];
  const float* W2   = (const float*)d_in[7];
  const float* b2   = (const float*)d_in[8];
  float* out = (float*)d_out;

  char* wptr = (char*)d_ws;
  auto carve = [&](size_t bytes) -> char* {
    char* p = wptr; wptr += (bytes + 255) & ~(size_t)255; return p;
  };
  __bf16* qb      = (__bf16*)carve((size_t)BATCH * NQ * DMODEL * 2);       //  4.2 MB
  __bf16* feat    = (__bf16*)carve((size_t)BATCH * APAD * 2 * DMODEL * 2); // 71.3 MB
  float*  v_area  = (float*) carve((size_t)BATCH * APAD * DMODEL * 4);     // 71.3 MB
  __bf16* hidden  = (__bf16*)carve((size_t)BATCH * APAD * DMODEL * 2);     // 35.7 MB
  __bf16* karea   = (__bf16*)carve((size_t)BATCH * APAD * DMODEL * 2);     // 35.7 MB
  __bf16* W1t     = (__bf16*)carve((size_t)256 * 512 * 2);
  __bf16* W2t     = (__bf16*)carve((size_t)256 * 256 * 2);
  float*  tb      = (float*) carve((size_t)9 * 256 * 4);
  __bf16* logitsb = (__bf16*)carve((size_t)BATCH * NQ * APAD * 2);         // 142.6 MB (L2-resident)

  cvt_q_kernel<<<(BATCH * NQ * DMODEL + 255) / 256, 256, 0, stream>>>(q, qb, (long)BATCH * NQ * DMODEL);
  w1t_kernel<<<(256 * 512 + 255) / 256, 256, 0, stream>>>(W1, W1t);
  w2t_kernel<<<(256 * 256 + 255) / 256, 256, 0, stream>>>(W2, W2t);
  typebias_kernel<<<(9 * 256 + 255) / 256, 256, 0, stream>>>(embh, embw, W1, b1, tb);

  long nA = (long)BATCH * APAD * (DMODEL / 4);
  area_feat_kernel<<<(unsigned)((nA + 255) / 256), 256, 0, stream>>>(k, v, feat, v_area);

  { // MLP layer 1: hidden = relu(feat @ W1[:512]^T + typebias[type])
    int mtiles = (BATCH * APAD) / 16, ngroups = 256 / 64;   // 4352, 4
    long waves = (long)mtiles * ngroups;
    gemm_wmma_kernel<512, 0><<<dim3((unsigned)((waves + 7) / 8), 1), 256, 0, stream>>>(
        feat, 2 * DMODEL, 0, W1t, 2 * DMODEL, 0, hidden, DMODEL, 0, mtiles, ngroups, tb);
  }
  { // MLP layer 2: k_area = hidden @ W2^T + b2
    int mtiles = (BATCH * APAD) / 16, ngroups = 256 / 64;
    long waves = (long)mtiles * ngroups;
    gemm_wmma_kernel<256, 1><<<dim3((unsigned)((waves + 7) / 8), 1), 256, 0, stream>>>(
        hidden, DMODEL, 0, W2t, DMODEL, 0, karea, DMODEL, 0, mtiles, ngroups, b2);
  }
  { // logits = q @ k_area^T  (batched over blockIdx.y)
    int mtiles = NQ / 16, ngroups = APAD / 64;              // 64, 136
    long waves = (long)mtiles * ngroups;                    // 8704
    gemm_wmma_kernel<256, 2><<<dim3((unsigned)((waves + 7) / 8), BATCH), 256, 0, stream>>>(
        qb, DMODEL, (long)NQ * DMODEL, karea, DMODEL, (long)APAD * DMODEL,
        logitsb, APAD, (long)NQ * APAD, mtiles, ngroups, nullptr);
  }
  select_out_kernel<<<BATCH * NQ / 4, 128, 0, stream>>>(logitsb, v_area, out);

  (void)in_sizes; (void)n_in; (void)out_size; (void)ws_size;
}
